// MultiHeadCrossAttention_33200097198636
// MI455X (gfx1250) — matrix operand order repaired and checked by reading, compiled, not run on hardware
//
#include <hip/hip_runtime.h>
#include <hip/hip_bf16.h>
#include <stdint.h>

// ---------------------------------------------------------------------------
// Problem constants (from reference): B=4, SQ=SKV=2048, E=1024, H=16, D=64
// ---------------------------------------------------------------------------
#define B_   4
#define SQ_  2048
#define SKV_ 2048
#define E_   1024
#define H_   16
#define D_   64

typedef __bf16 bf16_t;
typedef bf16_t v16bf __attribute__((ext_vector_type(16)));
typedef bf16_t v8bf  __attribute__((ext_vector_type(8)));
typedef float  v8f   __attribute__((ext_vector_type(8)));
typedef int    v4i   __attribute__((ext_vector_type(4)));

// gfx1250 async global->LDS path, if the toolchain exposes the builtins
#if defined(__has_builtin)
#if __has_builtin(__builtin_amdgcn_global_load_async_to_lds_b128) && \
    __has_builtin(__builtin_amdgcn_s_wait_asynccnt)
#define USE_ASYNC_LDS 1
#endif
#endif
#ifndef USE_ASYNC_LDS
#define USE_ASYNC_LDS 0
#endif

// fp32 -> bf16, round-to-nearest-even
__device__ __forceinline__ bf16_t f32_to_bf16(float f) {
    union { float f; uint32_t u; } c; c.f = f;
    uint32_t u = c.u;
    u += 0x7FFFu + ((u >> 16) & 1u);
    union { unsigned short s; bf16_t b; } r; r.s = (unsigned short)(u >> 16);
    return r.b;
}

// ---------------------------------------------------------------------------
// WMMA fragment loaders (layouts per CDNA5 ISA 7.12.2, wave32)
// A (16x32 bf16, MxK): lane L<16 -> row L, K {k0+0..7, k0+16..23};
//                      lane L>=16 -> row L-16, K {k0+8..15, k0+24..31}
// B (32x16 bf16, KxN): lane L<16 -> col L, K {k0+0..7, k0+16..23};
//                      lane L>=16 -> col L-16, K {k0+8..15, k0+24..31}
// C/D (16x16 f32):     VGPR r -> row r (lanes 0-15) / row 8+r (lanes 16-31),
//                      col = lane & 15
// ---------------------------------------------------------------------------
__device__ __forceinline__ v16bf load_A_bf(const bf16_t* p, int ld, int k0, int lane) {
    int r  = lane & 15;
    int ko = (lane & 16) ? 8 : 0;
    const bf16_t* q = p + (size_t)r * ld + k0 + ko;
    v8bf lo = *(const v8bf*)(q);
    v8bf hi = *(const v8bf*)(q + 16);
    v16bf out;
#pragma unroll
    for (int i = 0; i < 8; ++i) { out[i] = lo[i]; out[8 + i] = hi[i]; }
    return out;
}

__device__ __forceinline__ v16bf load_A_f32(const float* p, int ld, int k0, int lane) {
    int r  = lane & 15;
    int ko = (lane & 16) ? 8 : 0;
    const float* q = p + (size_t)r * ld + k0 + ko;
    v16bf out;
#pragma unroll
    for (int i = 0; i < 8; ++i) {
        out[i]     = f32_to_bf16(q[i]);
        out[8 + i] = f32_to_bf16(q[16 + i]);
    }
    return out;
}

__device__ __forceinline__ v16bf load_B_bf(const bf16_t* p, int ld, int k0, int lane) {
    int n  = lane & 15;
    int kb = (lane & 16) ? 8 : 0;
    const bf16_t* q = p + (size_t)n * ld + k0 + kb;
    v8bf lo = *(const v8bf*)(q);
    v8bf hi = *(const v8bf*)(q + 16);
    v16bf out;
#pragma unroll
    for (int i = 0; i < 8; ++i) { out[i] = lo[i]; out[8 + i] = hi[i]; }
    return out;
}

__device__ __forceinline__ v8f wmma_bf16(v16bf a, v16bf b, v8f c) {
    return __builtin_amdgcn_wmma_f32_16x16x32_bf16(
        /*neg_a=*/false, a, /*neg_b=*/false, b,
        /*c_mod=*/(short)0, c, /*reuse_a=*/false, /*reuse_b=*/false);
}

// 16-byte global -> LDS copy (async DMA when available)
__device__ __forceinline__ void stage16(const char* g, char* l) {
#if USE_ASYNC_LDS
    __builtin_amdgcn_global_load_async_to_lds_b128(
        (__attribute__((address_space(1))) v4i*)(g),
        (__attribute__((address_space(3))) v4i*)(l), 0, 0);
#else
    *(int4*)l = *(const int4*)g;
#endif
}

// ---------------------------------------------------------------------------
// Kernel 0: fp32 -> bf16 elementwise conversion (weights)
// ---------------------------------------------------------------------------
__global__ void convert_f32_bf16(const float* __restrict__ src,
                                 bf16_t* __restrict__ dst, int n) {
    int i = blockIdx.x * blockDim.x + threadIdx.x;
    if (i < n) dst[i] = f32_to_bf16(src[i]);
}

// ---------------------------------------------------------------------------
// Kernel 1: per-head Q/K/V projections.
// One wave per (kind, b, h, 16-row tile): 16x64 output = 4 N-tiles, K=64.
// Q -> Qb[b][h][s][d] (bf16), K -> Kb[b][h][s][d], V -> Vt[b][h][d][s]
// ---------------------------------------------------------------------------
__global__ __launch_bounds__(128)
void proj_kernel(const float* __restrict__ query, const float* __restrict__ keyv,
                 const bf16_t* __restrict__ wq, const bf16_t* __restrict__ wk,
                 const bf16_t* __restrict__ wv,
                 const float* __restrict__ bq, const float* __restrict__ bk,
                 const float* __restrict__ bv,
                 bf16_t* __restrict__ Qb, bf16_t* __restrict__ Kb,
                 bf16_t* __restrict__ Vt) {
    int wave = (blockIdx.x * blockDim.x + threadIdx.x) >> 5;
    int lane = threadIdx.x & 31;

    const int TILES = H_ * (SQ_ / 16);             // per (kind,b)
    int kind = wave / (B_ * TILES);                // 0=Q, 1=K, 2=V
    int rem  = wave % (B_ * TILES);
    int b    = rem / TILES;
    int h    = (rem / (SQ_ / 16)) % H_;
    int s0   = (rem % (SQ_ / 16)) * 16;

    const float* x = (kind == 0 ? query : keyv) +
                     ((size_t)b * SQ_ + s0) * E_ + (size_t)h * D_;
    const bf16_t* W   = (kind == 0 ? wq : (kind == 1 ? wk : wv)) + (size_t)h * D_ * D_;
    const float*  bia = (kind == 0 ? bq : (kind == 1 ? bk : bv)) + (size_t)h * D_;

    v16bf a0 = load_A_f32(x, E_, 0,  lane);
    v16bf a1 = load_A_f32(x, E_, 32, lane);

    v8f c[4];
#pragma unroll
    for (int t = 0; t < 4; ++t) {
        c[t] = v8f{};
        v16bf b0 = load_B_bf(W + (size_t)(t * 16) * D_, D_, 0,  lane);
        c[t] = wmma_bf16(a0, b0, c[t]);
        v16bf b1 = load_B_bf(W + (size_t)(t * 16) * D_, D_, 32, lane);
        c[t] = wmma_bf16(a1, b1, c[t]);
    }

    int col    = lane & 15;
    int rowoff = (lane & 16) ? 8 : 0;
#pragma unroll
    for (int t = 0; t < 4; ++t) {
        float bval = bia[t * 16 + col];
#pragma unroll
        for (int r = 0; r < 8; ++r) {
            float v = c[t][r] + bval;
            int row = r + rowoff;
            if (kind == 0) {
                Qb[(((size_t)b * H_ + h) * SQ_ + s0 + row) * D_ + t * 16 + col] = f32_to_bf16(v);
            } else if (kind == 1) {
                Kb[(((size_t)b * H_ + h) * SKV_ + s0 + row) * D_ + t * 16 + col] = f32_to_bf16(v);
            } else {
                Vt[(((size_t)b * H_ + h) * D_ + t * 16 + col) * SKV_ + s0 + row] = f32_to_bf16(v);
            }
        }
    }
}

// ---------------------------------------------------------------------------
// Kernel 2: flash attention.
// One block (4 waves) owns 64 Q rows of one (b,h); wave w takes rows
// [64*tile + 16*w, +16). K/V kv-blocks of 32 are staged into double-buffered
// LDS cooperatively (async DMA when available) and shared by all 4 waves:
// 4x less L2 traffic than per-wave streaming. Per kv-block per wave:
// 4 WMMAs for S, fp32 online softmax (shfl row reductions), P restaged
// through LDS to A-layout, 4 WMMAs for P*V.
// ---------------------------------------------------------------------------
#define KVB 32                       // kv block
#define NKVB (SKV_ / KVB)            // 64 iterations

__global__ __launch_bounds__(128)
void attn_kernel(const bf16_t* __restrict__ Qb, const bf16_t* __restrict__ Kb,
                 const bf16_t* __restrict__ Vt, bf16_t* __restrict__ Ctx) {
    // K tile: 32 kv-rows x 64 d (ld=64). V tile: 64 d-rows x 32 kv (ld=32).
    __shared__ __align__(16) bf16_t ldsK[2][KVB * D_];    // 2 x 4KB
    __shared__ __align__(16) bf16_t ldsV[2][D_ * KVB];    // 2 x 4KB
    __shared__ __align__(16) bf16_t ldsP[4][16 * 32];     // per-wave P staging

    int tid  = threadIdx.x;
    int w    = tid >> 5;
    int lane = tid & 31;

    int blk  = blockIdx.x;
    int b    = blk / (H_ * (SQ_ / 64));
    int h    = (blk / (SQ_ / 64)) % H_;
    int s0   = (blk % (SQ_ / 64)) * 64 + w * 16;

    const bf16_t* Qp = Qb + (((size_t)b * H_ + h) * SQ_ + s0) * D_;
    const bf16_t* Kp = Kb + (((size_t)b * H_ + h) * SKV_) * D_;
    const bf16_t* Vp = Vt + (((size_t)b * H_ + h) * D_) * SKV_;

    // --- cooperative stage of one kv block into LDS buffer `buf` ---
    // K block: 32*64 bf16 = 4KB contiguous. V block: 64 rows x 64B strided.
    // 128 threads x 2 chunks x 16B each per matrix.
    auto stage_block = [&](int buf, int kv) {
        const char* gk = (const char*)(Kp + (size_t)kv * D_);
        char* lk = (char*)&ldsK[buf][0];
        stage16(gk + tid * 16,          lk + tid * 16);
        stage16(gk + (tid + 128) * 16,  lk + (tid + 128) * 16);
#pragma unroll
        for (int half = 0; half < 2; ++half) {
            int c   = tid + half * 128;            // chunk id 0..255
            int row = c >> 2;                      // d row 0..63
            int off = (c & 3) * 16;                // byte offset in 64B row
            const char* gv = (const char*)(Vp + (size_t)row * SKV_ + kv) + off;
            char* lv = (char*)&ldsV[buf][0] + row * 64 + off;
            stage16(gv, lv);
        }
    };

    // Q fragments resident for the whole kv loop
    v16bf aq0 = load_A_bf(Qp, D_, 0,  lane);
    v16bf aq1 = load_A_bf(Qp, D_, 32, lane);

    v8f o[4];
#pragma unroll
    for (int t = 0; t < 4; ++t) o[t] = v8f{};
    float m[8], l[8];
#pragma unroll
    for (int r = 0; r < 8; ++r) { m[r] = -3.0e38f; l[r] = 0.0f; }

    bf16_t* P = &ldsP[w][0];
    const int col    = lane & 15;
    const int rowoff = (lane & 16) ? 8 : 0;
    const float scale = 0.125f;                    // 1/sqrt(64)

    stage_block(0, 0);

    for (int i = 0; i < NKVB; ++i) {
        int cur = i & 1;
        // all waves finished reading buf cur^1 (iteration i-1) before overwrite
        __syncthreads();
        if (i + 1 < NKVB) {
            stage_block(cur ^ 1, (i + 1) * KVB);
#if USE_ASYNC_LDS
            __builtin_amdgcn_s_wait_asynccnt(4);   // drain stage(cur), keep next in flight
#endif
        } else {
#if USE_ASYNC_LDS
            __builtin_amdgcn_s_wait_asynccnt(0);
#endif
        }
        // everyone's stage(cur) visible
        __syncthreads();

        const bf16_t* Kt = &ldsK[cur][0];
        const bf16_t* Vl = &ldsV[cur][0];

        // ---- S = Q K^T : 16x32 score block (two 16x16 tiles) ----
        v8f s0v = v8f{};
        v8f s1v = v8f{};
        s0v = wmma_bf16(aq0, load_B_bf(Kt,             D_, 0,  lane), s0v);
        s0v = wmma_bf16(aq1, load_B_bf(Kt,             D_, 32, lane), s0v);
        s1v = wmma_bf16(aq0, load_B_bf(Kt + 16 * D_,   D_, 0,  lane), s1v);
        s1v = wmma_bf16(aq1, load_B_bf(Kt + 16 * D_,   D_, 32, lane), s1v);

        // ---- online softmax, per row r (rows live across 16 lanes) ----
#pragma unroll
        for (int r = 0; r < 8; ++r) {
            float x0 = s0v[r] * scale;
            float x1 = s1v[r] * scale;
            float mx = fmaxf(x0, x1);
#pragma unroll
            for (int off = 8; off > 0; off >>= 1)
                mx = fmaxf(mx, __shfl_xor(mx, off, 16));
            float mnew = fmaxf(m[r], mx);
            float corr = __expf(m[r] - mnew);
            float p0 = __expf(x0 - mnew);
            float p1 = __expf(x1 - mnew);
            float rs = p0 + p1;
#pragma unroll
            for (int off = 8; off > 0; off >>= 1)
                rs += __shfl_xor(rs, off, 16);
            l[r] = l[r] * corr + rs;
            m[r] = mnew;
            o[0][r] *= corr; o[1][r] *= corr; o[2][r] *= corr; o[3][r] *= corr;
            P[(r + rowoff) * 32 + col]      = f32_to_bf16(p0);
            P[(r + rowoff) * 32 + 16 + col] = f32_to_bf16(p1);
        }

        // intra-wave DS ordering: P stores visible to the reads below
        asm volatile("s_wait_dscnt 0" ::: "memory");

        // ---- O += P * V : A = P (16x32), B = V from LDS tile ----
        v16bf ap = load_A_bf(P, 32, 0, lane);
#pragma unroll
        for (int t = 0; t < 4; ++t) {
            v16bf bv = load_B_bf(Vl + (size_t)(t * 16) * KVB, KVB, 0, lane);
            o[t] = wmma_bf16(ap, bv, o[t]);
        }
    }

    // ---- normalize and write ctx (bf16, [B*SQ, E] row-major) ----
#pragma unroll
    for (int r = 0; r < 8; ++r) {
        float inv = 1.0f / l[r];
        int row = s0 + r + rowoff;
#pragma unroll
        for (int t = 0; t < 4; ++t) {
            Ctx[((size_t)b * SQ_ + row) * E_ + (size_t)h * D_ + t * 16 + col] =
                f32_to_bf16(o[t][r] * inv);
        }
    }
}

// ---------------------------------------------------------------------------
// Kernel 3: output projection  out = ctx @ wo^T + bo   (8192 x 1024 x 1024)
// One wave per (16-row stripe, 64-col stripe): 4 N-tiles, K loop over 1024.
// ---------------------------------------------------------------------------
__global__ __launch_bounds__(128)
void outproj_kernel(const bf16_t* __restrict__ Ctx, const bf16_t* __restrict__ Wo,
                    const float* __restrict__ bo, float* __restrict__ out) {
    int wave = (blockIdx.x * blockDim.x + threadIdx.x) >> 5;
    int lane = threadIdx.x & 31;

    int mt = wave / (E_ / 64);
    int nt = wave % (E_ / 64);
    int m0 = mt * 16;
    int n0 = nt * 64;

    v8f c[4];
#pragma unroll
    for (int t = 0; t < 4; ++t) c[t] = v8f{};

    const bf16_t* Arow = Ctx + (size_t)m0 * E_;
    for (int k0 = 0; k0 < E_; k0 += 32) {
        v16bf a = load_A_bf(Arow, E_, k0, lane);
#pragma unroll
        for (int t = 0; t < 4; ++t) {
            v16bf bw = load_B_bf(Wo + (size_t)(n0 + t * 16) * E_, E_, k0, lane);
            c[t] = wmma_bf16(a, bw, c[t]);
        }
    }

    int col    = lane & 15;
    int rowoff = (lane & 16) ? 8 : 0;
#pragma unroll
    for (int t = 0; t < 4; ++t) {
        float bval = bo[n0 + t * 16 + col];
#pragma unroll
        for (int r = 0; r < 8; ++r) {
            out[((size_t)m0 + r + rowoff) * E_ + n0 + t * 16 + col] = c[t][r] + bval;
        }
    }
}

// ---------------------------------------------------------------------------
// Launch
// ---------------------------------------------------------------------------
extern "C" void kernel_launch(void* const* d_in, const int* in_sizes, int n_in,
                              void* d_out, int out_size, void* d_ws, size_t ws_size,
                              hipStream_t stream) {
    const float* query = (const float*)d_in[0];
    const float* keyv  = (const float*)d_in[1];
    const float* wq    = (const float*)d_in[2];
    const float* bq    = (const float*)d_in[3];
    const float* wk    = (const float*)d_in[4];
    const float* bk    = (const float*)d_in[5];
    const float* wv    = (const float*)d_in[6];
    const float* bv    = (const float*)d_in[7];
    const float* wo    = (const float*)d_in[8];
    const float* bo    = (const float*)d_in[9];
    float* out = (float*)d_out;

    // workspace layout (bf16 elements)
    const size_t N_WHEAD = (size_t)H_ * D_ * D_;           //  65536 per weight
    const size_t N_WO    = (size_t)E_ * E_;                // 1048576
    const size_t N_QKV   = (size_t)B_ * H_ * SQ_ * D_;     // 8388608

    bf16_t* base  = (bf16_t*)d_ws;
    bf16_t* wq_bf = base;
    bf16_t* wk_bf = wq_bf + N_WHEAD;
    bf16_t* wv_bf = wk_bf + N_WHEAD;
    bf16_t* wo_bf = wv_bf + N_WHEAD;
    bf16_t* Qb    = wo_bf + N_WO;
    bf16_t* Kb    = Qb + N_QKV;
    bf16_t* Vt    = Kb + N_QKV;
    bf16_t* Ctx   = Vt + N_QKV;

    // 0) weight down-conversion
    convert_f32_bf16<<<(int)((N_WHEAD + 255) / 256), 256, 0, stream>>>(wq, wq_bf, (int)N_WHEAD);
    convert_f32_bf16<<<(int)((N_WHEAD + 255) / 256), 256, 0, stream>>>(wk, wk_bf, (int)N_WHEAD);
    convert_f32_bf16<<<(int)((N_WHEAD + 255) / 256), 256, 0, stream>>>(wv, wv_bf, (int)N_WHEAD);
    convert_f32_bf16<<<(int)((N_WO    + 255) / 256), 256, 0, stream>>>(wo, wo_bf, (int)N_WO);

    // 1) Q/K/V projections: 3 * B*H*(SQ/16) = 24576 waves, 4 waves/block
    proj_kernel<<<6144, 128, 0, stream>>>(query, keyv, wq_bf, wk_bf, wv_bf,
                                          bq, bk, bv, Qb, Kb, Vt);

    // 2) flash attention: B*H*(SQ/64) = 2048 blocks of 4 waves
    attn_kernel<<<2048, 128, 0, stream>>>(Qb, Kb, Vt, Ctx);

    // 3) output projection: (B*SQ/16)*(E/64) = 8192 waves, 4 waves/block
    outproj_kernel<<<2048, 128, 0, stream>>>(Ctx, wo_bf, bo, out);
}
